// SelfAttentionLayer_15410342658657
// MI455X (gfx1250) — compile-verified, matrix operand order checked by reading, not run on hardware
//
#include <hip/hip_runtime.h>
#include <hip/hip_bf16.h>

// ---------------------------------------------------------------------------
// Self-attention, N=8192, C=1024, fp32 in/out, bf16 WMMA compute (gfx1250).
//
//   q = x@wq^T + bq ; k = x@wk^T + bk ; v = x@wv^T + bv
//   S = q@k^T (+const shift -> cancels in softmax) ; P = softmax(S) ; O = P@v
//
// Compute-bound (~326 GFLOP vs <1GB traffic @ 23.3TB/s): all matmuls run on
// v_wmma_f32_16x16x32_bf16. Tiles are staged into LDS by the Tensor Data
// Mover (tensor_load_to_lds, TENSORcnt), double-buffered, so global traffic
// is shared by all 4 waves of the block and overlapped with WMMA issue.
// ---------------------------------------------------------------------------

typedef __attribute__((ext_vector_type(16))) __bf16 v16bf;
typedef __attribute__((ext_vector_type(8)))  __bf16 v8bf;
typedef __attribute__((ext_vector_type(8)))  float  v8f;
typedef __attribute__((ext_vector_type(4)))  unsigned int v4u;
typedef __attribute__((ext_vector_type(8)))  int    v8i;
typedef __attribute__((ext_vector_type(4)))  int    v4i;

#define NTOK 8192
#define CDIM 1024

__device__ __forceinline__ __bf16 f2bf(float f) {
  union { float f; unsigned u; } v; v.f = f;
  unsigned r = (v.u + 0x7FFFu + ((v.u >> 16) & 1u)) >> 16;  // RNE
  union { unsigned short s; __bf16 b; } o; o.s = (unsigned short)r;
  return o.b;
}

// Low 32 bits of a flat pointer to __shared__ == LDS byte offset (aperture
// mapping discards the upper bits; see CDNA5 ISA).
__device__ __forceinline__ unsigned lds_addr32(const void* p) {
  return (unsigned)(unsigned long long)p;
}

// TDM: async-load a 32(elem)x128(row) bf16 tile (row stride = stride_elems)
// from global into contiguous LDS (128 rows x 64B). D# layout per
// CDNA5 ISA (2D: upper descriptor groups zero).
// This toolchain exposes the 6-arg builtin:
//   (uint32x4 g0, int32x8 g1, int32x4, int32x4, int32x8, i32 cpol)
__device__ __forceinline__ void tdm_load_tile_32x128(
    unsigned lds_byte_off, const __bf16* gsrc, int stride_elems) {
  unsigned long long ga = (unsigned long long)(const void*)gsrc;
  v4u g0;
  g0.x = 1u;                                    // count=1, user descriptor
  g0.y = lds_byte_off;                          // lds_addr (bytes)
  g0.z = (unsigned)ga;                          // global_addr[31:0]
  g0.w = ((unsigned)(ga >> 32) & 0x01FFFFFFu)   // global_addr[56:32]
         | (2u << 30);                          // type = 2
  v8i g1;
  g1[0] = (int)(1u << 16);                      // data_size=1 -> 2 bytes
  g1[1] = (int)(32u << 16);                     // tensor_dim0 = 32 (lo16)
  g1[2] = (int)(128u << 16);                    // dim0 hi=0 | tensor_dim1=128
  g1[3] = (int)(32u << 16);                     // dim1 hi=0 | tile_dim0 = 32
  g1[4] = 128;                                  // tile_dim1 = 128 (tile_dim2=0)
  g1[5] = (int)(unsigned)stride_elems;          // tensor_dim0_stride lo32
  g1[6] = 0;                                    // stride hi / dim1_stride
  g1[7] = 0;
  v4i z4 = {0, 0, 0, 0};
  v8i z8 = {0, 0, 0, 0, 0, 0, 0, 0};
  __builtin_amdgcn_tensor_load_to_lds(g0, g1, z4, z4, z8, 0);
}

// ---------------------------------------------------------------- convert ---
__global__ void cvt_f32_to_bf16(const float* __restrict__ src,
                                __bf16* __restrict__ dst, long long n) {
  long long i = (long long)blockIdx.x * blockDim.x + threadIdx.x;
  long long stride = (long long)gridDim.x * blockDim.x;
  for (; i < n; i += stride) dst[i] = f2bf(src[i]);
}

// ------------------------------------------------------------------- GEMM ---
// C[m][n] = sum_k A[m][k] * B[n][k]   (A,B bf16 row-major)
// Block = 4 waves -> 128x128 tile; wave tile 64x64 (4x4 of 16x16x32 WMMA).
// Per 32-wide K step, wave 0 TDM-loads the next 128x32 A and B tiles into the
// spare LDS buffer while all waves compute from the current one.
// Fragment chunking per ISA 16-bit A layout: lane L (lm=L&15, hs=L>>4) holds
// row (base+lm), K chunks [hs*8..+7] and [16+hs*8..+7] (two ds_load_b128).
// EPI 0: +bias, bf16 store out[m*ldo+n]
// EPI 1: +bias, bf16 store out[n*ldo+m]   (transposed, for V^T)
// EPI 2: f32 store out[m*ldo+n]           (scores S / final output O)
template <int EPI>
__global__ __launch_bounds__(128)
void gemm_bf16_wmma(const __bf16* __restrict__ A, const __bf16* __restrict__ B,
                    const float* __restrict__ bias, void* __restrict__ outp,
                    int Kk, int lda, int ldb, int ldo) {
  __shared__ __align__(128) __bf16 tileA[2][128 * 32];
  __shared__ __align__(128) __bf16 tileB[2][128 * 32];

  const int lane = threadIdx.x & 31;
  const int wave = threadIdx.x >> 5;
  const int mblk = blockIdx.x * 128;
  const int nblk = blockIdx.y * 128;
  const int mloc = (wave >> 1) * 64;   // wave's rows within the block tile
  const int nloc = (wave & 1) * 64;    // wave's cols within the block tile
  const int lm = lane & 15;
  const int hs = (lane >> 4) & 1;

  v8f acc[4][4];
  const v8f vzero = {0.f, 0.f, 0.f, 0.f, 0.f, 0.f, 0.f, 0.f};
#pragma unroll
  for (int i = 0; i < 4; ++i)
#pragma unroll
    for (int j = 0; j < 4; ++j) acc[i][j] = vzero;

  const bool issuer = (wave == 0);
  if (issuer) {
    tdm_load_tile_32x128(lds_addr32(&tileA[0][0]),
                         A + (size_t)mblk * lda, lda);
    tdm_load_tile_32x128(lds_addr32(&tileB[0][0]),
                         B + (size_t)nblk * ldb, ldb);
  }

  const int nsteps = Kk / 32;
  for (int t = 0; t < nsteps; ++t) {
    const int buf = t & 1;
    if (issuer) {
      if (t + 1 < nsteps) {
        const int k0n = (t + 1) * 32;
        tdm_load_tile_32x128(lds_addr32(&tileA[buf ^ 1][0]),
                             A + (size_t)mblk * lda + k0n, lda);
        tdm_load_tile_32x128(lds_addr32(&tileB[buf ^ 1][0]),
                             B + (size_t)nblk * ldb + k0n, ldb);
        __builtin_amdgcn_s_wait_tensorcnt(2);  // current buffer's 2 loads done
      } else {
        __builtin_amdgcn_s_wait_tensorcnt(0);
      }
    }
    __syncthreads();  // buffer `buf` valid for all waves

    v16bf af[4], bfv[4];
#pragma unroll
    for (int i = 0; i < 4; ++i) {
      const __bf16* p = &tileA[buf][(mloc + 16 * i + lm) * 32 + hs * 8];
      v8bf lo = *(const v8bf*)p;
      v8bf hi = *(const v8bf*)(p + 16);
      af[i] = __builtin_shufflevector(lo, hi, 0, 1, 2, 3, 4, 5, 6, 7,
                                      8, 9, 10, 11, 12, 13, 14, 15);
    }
#pragma unroll
    for (int j = 0; j < 4; ++j) {
      const __bf16* p = &tileB[buf][(nloc + 16 * j + lm) * 32 + hs * 8];
      v8bf lo = *(const v8bf*)p;
      v8bf hi = *(const v8bf*)(p + 16);
      bfv[j] = __builtin_shufflevector(lo, hi, 0, 1, 2, 3, 4, 5, 6, 7,
                                       8, 9, 10, 11, 12, 13, 14, 15);
    }
#pragma unroll
    for (int i = 0; i < 4; ++i)
#pragma unroll
      for (int j = 0; j < 4; ++j)
        acc[i][j] = __builtin_amdgcn_wmma_f32_16x16x32_bf16(
            false, af[i], false, bfv[j], (short)0, acc[i][j], false, false);

    __syncthreads();  // all waves done reading `buf` before it is re-filled
  }

  // Epilogue. D layout: lane holds col n = nbase+16j+lm; VGPR r holds row
  // m = mbase+16i+8*hs+r.
#pragma unroll
  for (int i = 0; i < 4; ++i) {
#pragma unroll
    for (int j = 0; j < 4; ++j) {
      const int col = nblk + nloc + 16 * j + lm;
      const int m0 = mblk + mloc + 16 * i + 8 * hs;
      float bv = 0.f;
      if (EPI != 2) bv = bias[col];
#pragma unroll
      for (int r = 0; r < 8; ++r) {
        const float val = acc[i][j][r] + bv;
        const int m = m0 + r;
        if (EPI == 0) {
          ((__bf16*)outp)[(size_t)m * ldo + col] = f2bf(val);
        } else if (EPI == 1) {
          ((__bf16*)outp)[(size_t)col * ldo + m] = f2bf(val);
        } else {
          ((float*)outp)[(size_t)m * ldo + col] = val;
        }
      }
    }
  }
}

// ---------------------------------------------------------------- softmax ---
// One block per row of S (f32, length 8192). Shift constant cancels here.
__global__ __launch_bounds__(256)
void softmax_rows(const float* __restrict__ S, __bf16* __restrict__ P, int Nn) {
  __shared__ float red[256];
  const int row = blockIdx.x;
  const int tid = threadIdx.x;
  const float* s = S + (size_t)row * Nn;

  float mx = -3.4e38f;
  for (int j = tid; j < Nn; j += 256) mx = fmaxf(mx, s[j]);
  red[tid] = mx; __syncthreads();
  for (int w = 128; w > 0; w >>= 1) {
    if (tid < w) red[tid] = fmaxf(red[tid], red[tid + w]);
    __syncthreads();
  }
  mx = red[0]; __syncthreads();

  float sum = 0.f;
  for (int j = tid; j < Nn; j += 256) sum += __expf(s[j] - mx);
  red[tid] = sum; __syncthreads();
  for (int w = 128; w > 0; w >>= 1) {
    if (tid < w) red[tid] += red[tid + w];
    __syncthreads();
  }
  const float inv = 1.f / red[0];

  __bf16* p = P + (size_t)row * Nn;
  for (int j = tid; j < Nn; j += 256) p[j] = f2bf(__expf(s[j] - mx) * inv);
}

// ----------------------------------------------------------------- launch ---
extern "C" void kernel_launch(void* const* d_in, const int* in_sizes, int n_in,
                              void* d_out, int out_size, void* d_ws, size_t ws_size,
                              hipStream_t stream) {
  const float* x  = (const float*)d_in[0];
  const float* wq = (const float*)d_in[1];
  const float* bq = (const float*)d_in[2];
  const float* wk = (const float*)d_in[3];
  const float* bk = (const float*)d_in[4];
  const float* wv = (const float*)d_in[5];
  const float* bv = (const float*)d_in[6];
  float* out = (float*)d_out;

  const size_t N = NTOK, C = CDIM;
  char* ws = (char*)d_ws;
  size_t off = 0;
  auto take = [&](size_t bytes) {
    char* p = ws + off;
    off = (off + bytes + 255) & ~(size_t)255;
    return p;
  };
  __bf16* Xb  = (__bf16*)take(N * C * 2);   // x in bf16
  __bf16* Wqb = (__bf16*)take(C * C * 2);
  __bf16* Wkb = (__bf16*)take(C * C * 2);
  __bf16* Wvb = (__bf16*)take(C * C * 2);
  __bf16* Qb  = (__bf16*)take(N * C * 2);   // q bf16 [N][C]
  __bf16* Kb  = (__bf16*)take(N * C * 2);   // k bf16 [N][C]
  __bf16* Vtb = (__bf16*)take(N * C * 2);   // v^T bf16 [C][N]
  float*  Sf  = (float*) take(N * N * 4);   // scores f32 [N][N]
  __bf16* Pb  = (__bf16*)take(N * N * 2);   // softmax bf16 [N][N]
  (void)ws_size;

  // 1) fp32 -> bf16 conversions
  cvt_f32_to_bf16<<<2048, 256, 0, stream>>>(x,  Xb,  (long long)(N * C));
  cvt_f32_to_bf16<<<1024, 256, 0, stream>>>(wq, Wqb, (long long)(C * C));
  cvt_f32_to_bf16<<<1024, 256, 0, stream>>>(wk, Wkb, (long long)(C * C));
  cvt_f32_to_bf16<<<1024, 256, 0, stream>>>(wv, Wvb, (long long)(C * C));

  // 2) projections: q,k row-major bf16 (+bias); v stored transposed [C][N]
  dim3 gproj(N / 128, C / 128);  // 64 x 8
  gemm_bf16_wmma<0><<<gproj, 128, 0, stream>>>(Xb, Wqb, bq, Qb,
                                               (int)C, (int)C, (int)C, (int)C);
  gemm_bf16_wmma<0><<<gproj, 128, 0, stream>>>(Xb, Wkb, bk, Kb,
                                               (int)C, (int)C, (int)C, (int)C);
  gemm_bf16_wmma<1><<<gproj, 128, 0, stream>>>(Xb, Wvb, bv, Vtb,
                                               (int)C, (int)C, (int)C, (int)N);

  // 3) scores: S = Q @ K^T  (both row-major [N][C], B rows = K rows)
  dim3 gscore(N / 128, N / 128);  // 64 x 64
  gemm_bf16_wmma<2><<<gscore, 128, 0, stream>>>(Qb, Kb, nullptr, Sf,
                                                (int)C, (int)C, (int)C, (int)N);

  // 4) row softmax (constant logit shift cancels)
  softmax_rows<<<N, 256, 0, stream>>>(Sf, Pb, (int)N);

  // 5) O = P @ V : A = P [N][N], B = V^T [C][N] -> contiguous per output col
  dim3 gout(N / 128, C / 128);  // 64 x 8
  gemm_bf16_wmma<2><<<gout, 128, 0, stream>>>(Pb, Vtb, nullptr, out,
                                              (int)N, (int)N, (int)N, (int)C);
  (void)in_sizes; (void)n_in; (void)out_size;
}
